// MPSRNN1D_local_75342316307001
// MI455X (gfx1250) — matrix operand order repaired and checked by reading, not compiled, not run on hardware
//
#include <hip/hip_runtime.h>

typedef __attribute__((ext_vector_type(16))) _Float16 v16h;
typedef __attribute__((ext_vector_type(8)))  _Float16 v8h;
typedef __attribute__((ext_vector_type(8)))  float    v8f;

constexpr int kL = 64;        // sites
constexpr int kS = 2;         // local dim
constexpr int kB = 32;        // bond dim (== WMMA K)
constexpr int kWaves = 8;     // waves per block
constexpr int kSPW = 16;      // samples per wave (== WMMA N)
constexpr int kSPB = kWaves * kSPW;  // 128 samples per block
constexpr float kEPS = 1e-7f;

constexpr int kMElems   = kS * kB * kB;        // 2048 complex elems per site
constexpr int kWHalves  = 2 * kMElems;         // 4096 f16 per site (re+im planes)
constexpr int kWBytes   = kWHalves * 2;        // 8192 bytes per site
constexpr int kWChunks  = kWBytes / 16;        // 512 b128 chunks per site

// ---------------------------------------------------------------------------
// One-time pass: convert complex64 M to per-site f16 planes [site][comp][s*B*B]
// laid out exactly as the main kernel's LDS A-fragment image.
// ---------------------------------------------------------------------------
__global__ void convert_weights(const float2* __restrict__ Mg,
                                _Float16* __restrict__ W) {
  int idx = blockIdx.x * blockDim.x + threadIdx.x;
  if (idx >= kL * kMElems) return;
  int site = idx / kMElems;
  int k    = idx % kMElems;
  float2 m = Mg[idx];
  _Float16* dst = W + (size_t)site * kWHalves;
  dst[k]           = (_Float16)m.x;   // re plane
  dst[kMElems + k] = (_Float16)m.y;   // im plane
}

// ---------------------------------------------------------------------------
// Main recurrence kernel.
// ---------------------------------------------------------------------------
__global__ __launch_bounds__(256, 2)
void mpsrnn_fwd(const int* __restrict__ inp,      // (BATCH, L) int32
                const float2* __restrict__ Mg,    // (L, S, B, B) complex64
                const float2* __restrict__ Vg,    // (L, S, B) complex64
                const float*  __restrict__ LG,    // (L, B) float32
                const _Float16* __restrict__ Wf,  // preconverted f16 weights (or null)
                int useAsync,
                float* __restrict__ out,          // (BATCH,) float32
                int batch)
{
  // Double-buffered site weights (f16 planes): [buf][comp][s][a][b]
  __shared__ __align__(32) _Float16 sM[2][2][kS][kB][kB];   // 2 x 8 KB
  __shared__ __align__(32) float    sVre[kS * kB];
  __shared__ __align__(32) float    sVim[kS * kB];
  __shared__ __align__(32) float    sEG[kB];
  // Recurrent hv per wave, laid out [sample][a] so B-fragment reads are contiguous
  __shared__ __align__(32) _Float16 sHVre[kWaves][kSPW][kB]; // 8 KB
  __shared__ __align__(32) _Float16 sHVim[kWaves][kSPW][kB]; // 8 KB

  const int tid  = threadIdx.x;
  const int wave = tid >> 5;
  const int lane = tid & 31;
  const int n    = lane & 15;   // sample column within tile (WMMA N index)
  const int hi   = lane >> 4;   // lane-group (selects K/M halves per ISA layouts)
  const int kb   = hi * 8;
  const int sample = blockIdx.x * kSPB + wave * kSPW + n;
  const bool valid = sample < batch;

  // Issue the async DMA of one site's f16 weights into LDS buffer `buf`.
  // 512 b128 chunks / 256 threads = 2 chunks each; lane-contiguous for coalescing.
  auto issue_site = [&](int site, int buf) {
    unsigned ldsBase = (unsigned)(uintptr_t)(&sM[buf][0][0][0][0]);
    const _Float16* gbase = Wf + (size_t)site * kWHalves;   // uniform -> SGPR pair
    #pragma unroll
    for (int c = 0; c < 2; ++c) {
      int idx = tid + c * 256;                // chunk 0..511
      unsigned lds  = ldsBase + idx * 16;
      unsigned voff = idx * 16;
      asm volatile("global_load_async_to_lds_b128 %0, %1, %2 offset:0"
                   :: "v"(lds), "v"(voff), "s"(gbase)
                   : "memory");
    }
  };

  if (useAsync) issue_site(0, 0);

  float logpsi = 0.f;
  int counts0 = 0, counts1 = 0;
  int q_prev = 0;

  for (int i = 0; i < kL; ++i) {
    const int buf = useAsync ? (i & 1) : 0;
    __syncthreads();  // readers of the buffer we are about to refill are done

    if (useAsync) {
      if (i + 1 < kL) issue_site(i + 1, (i + 1) & 1);   // overlaps this site's math
    } else {
      // Fallback staging: f32 -> f16 conversion in-loop.
      _Float16* flatM = &sM[0][0][0][0][0];
      #pragma unroll
      for (int k = tid; k < kMElems; k += 256) {
        float2 m = Mg[i * kMElems + k];
        flatM[k]           = (_Float16)m.x;
        flatM[kMElems + k] = (_Float16)m.y;
      }
      if (i + 1 < kL)
        __builtin_prefetch(&Mg[(i + 1) * kMElems + tid * 8], 0, 1);
    }

    // Small per-site tensors via regular loads.
    if (tid < kS * kB) {
      float2 vv = Vg[i * kS * kB + tid];
      sVre[tid] = vv.x;
      sVim[tid] = vv.y;
    }
    if (tid < kB) sEG[tid] = __expf(LG[i * kB + tid]);

    if (useAsync) {
      // Async loads complete in issue order: waiting ASYNCcnt<=2 guarantees
      // this site's 2 chunks landed while next site's 2 stay in flight.
      if (i + 1 < kL) asm volatile("s_wait_asynccnt 0x2" ::: "memory");
      else            asm volatile("s_wait_asynccnt 0x0" ::: "memory");
    }
    __syncthreads();  // publish site-i weights + v/eg to all waves

    // current-site quantum number for this lane's sample
    int q_cur = valid ? inp[sample * kL + i] : 0;

    // ---- B fragments (32x16 f16): K runs {kb..kb+7}, {16+kb..16+kb+7}, N = lane&15 ----
    v16h Bre, Bim;
    if (i == 0) {
      #pragma unroll
      for (int e = 0; e < 16; ++e) { Bre[e] = (_Float16)1.0f; Bim[e] = (_Float16)0.0f; }
    } else {
      Bre = __builtin_shufflevector(*(const v8h*)&sHVre[wave][n][kb], *(const v8h*)&sHVre[wave][n][16 + kb], 0,1,2,3,4,5,6,7,8,9,10,11,12,13,14,15);
      Bim = __builtin_shufflevector(*(const v8h*)&sHVim[wave][n][kb], *(const v8h*)&sHVim[wave][n][16 + kb], 0,1,2,3,4,5,6,7,8,9,10,11,12,13,14,15);
    }
    v16h Bnim = -Bim;

    // ---- complex GEMM via 16 chained WMMAs, C preloaded with bias v ----
    v8f cre[kS][2], cim[kS][2];
    #pragma unroll
    for (int s = 0; s < kS; ++s) {
      #pragma unroll
      for (int t = 0; t < 2; ++t) {
        // A fragment 16x32 f16: row m = lane&15; K runs {kb..kb+7}, {16+kb..16+kb+7}
        v8h r0 = *(const v8h*)&sM[buf][0][s][t * 16 + n][kb];
        v8h r1 = *(const v8h*)&sM[buf][0][s][t * 16 + n][16 + kb];
        v16h Are = __builtin_shufflevector(r0, r1, 0,1,2,3,4,5,6,7,8,9,10,11,12,13,14,15);
        r0 = *(const v8h*)&sM[buf][1][s][t * 16 + n][kb];
        r1 = *(const v8h*)&sM[buf][1][s][t * 16 + n][16 + kb];
        v16h Aim = __builtin_shufflevector(r0, r1, 0,1,2,3,4,5,6,7,8,9,10,11,12,13,14,15);

        // C/D element r -> row a = t*16 + 8*hi + r, col = sample (v broadcast)
        v8f cr = *(const v8f*)&sVre[s * kB + t * 16 + kb];
        v8f ci = *(const v8f*)&sVim[s * kB + t * 16 + kb];

        cr = __builtin_amdgcn_wmma_f32_16x16x32_f16(false, Are, false, Bre,  (short)0, cr, false, false);
        cr = __builtin_amdgcn_wmma_f32_16x16x32_f16(false, Aim, false, Bnim, (short)0, cr, false, false);
        ci = __builtin_amdgcn_wmma_f32_16x16x32_f16(false, Are, false, Bim,  (short)0, ci, false, false);
        ci = __builtin_amdgcn_wmma_f32_16x16x32_f16(false, Aim, false, Bre,  (short)0, ci, false, false);
        cre[s][t] = cr;
        cim[s][t] = ci;
      }
    }

    // ---- per-sample reductions: norm + gamma-weighted probabilities ----
    v8f egv0 = *(const v8f*)&sEG[kb];
    v8f egv1 = *(const v8f*)&sEG[16 + kb];
    float sumsq = 0.f, ps0 = 0.f, ps1 = 0.f;
    #pragma unroll
    for (int t = 0; t < 2; ++t) {
      #pragma unroll
      for (int r = 0; r < 8; ++r) {
        float a0 = cre[0][t][r], b0 = cim[0][t][r];
        float a1 = cre[1][t][r], b1 = cim[1][t][r];
        float m0 = a0 * a0 + b0 * b0;
        float m1 = a1 * a1 + b1 * b1;
        sumsq += m0 + m1;
        float g = (t == 0) ? egv0[r] : egv1[r];
        ps0 += m0 * g;
        ps1 += m1 * g;
      }
    }
    // combine the two lanes holding this sample (rows split by hi)
    sumsq += __shfl_xor(sumsq, 16, 32);
    ps0   += __shfl_xor(ps0,   16, 32);
    ps1   += __shfl_xor(ps1,   16, 32);

    float invmean = (float)(kS * kB) / sumsq;   // 1 / mean|h|^2
    float rinv    = sqrtf(invmean);             // normalization factor for h
    float p0 = ps0 * invmean;
    float p1 = ps1 * invmean;

    if (i > 0) { counts0 += (q_prev == 0); counts1 += (q_prev == 1); }
    if (counts0 >= kL / 2) p0 = kEPS;
    if (counts1 >= kL / 2) p1 = kEPS;
    float psel = ((q_cur == 0) ? p0 : p1) / (p0 + p1);
    logpsi += 0.5f * __logf(psel);

    // ---- write normalized hv (selected by q_cur) for next site's B fragment ----
    #pragma unroll
    for (int t = 0; t < 2; ++t) {
      v8f hr = (q_cur == 0) ? cre[0][t] : cre[1][t];
      v8f hm = (q_cur == 0) ? cim[0][t] : cim[1][t];
      v8h hr16, hm16;
      #pragma unroll
      for (int r = 0; r < 8; ++r) {
        hr16[r] = (_Float16)(hr[r] * rinv);
        hm16[r] = (_Float16)(hm[r] * rinv);
      }
      *(v8h*)&sHVre[wave][n][t * 16 + kb] = hr16;
      *(v8h*)&sHVim[wave][n][t * 16 + kb] = hm16;
    }
    q_prev = q_cur;
  }

  if (hi == 0 && valid) out[sample] = logpsi;
}

extern "C" void kernel_launch(void* const* d_in, const int* in_sizes, int n_in,
                              void* d_out, int out_size, void* d_ws, size_t ws_size,
                              hipStream_t stream) {
  (void)n_in; (void)out_size;
  const int*    inp = (const int*)d_in[0];
  const float2* Mg  = (const float2*)d_in[1];   // complex64 -> interleaved float2
  const float2* Vg  = (const float2*)d_in[2];
  const float*  LG  = (const float*)d_in[3];
  float* out = (float*)d_out;

  const size_t needW = (size_t)kL * kWBytes;    // 512 KB of preconverted f16
  const int useAsync = (d_ws != nullptr && ws_size >= needW) ? 1 : 0;
  _Float16* Wf = (_Float16*)d_ws;

  if (useAsync) {
    int total = kL * kMElems;
    convert_weights<<<(total + 255) / 256, 256, 0, stream>>>(Mg, Wf);
  }

  int batch  = in_sizes[0] / kL;                // inputs is (BATCH, L)
  int blocks = (batch + kSPB - 1) / kSPB;
  mpsrnn_fwd<<<blocks, 256, 0, stream>>>(inp, Mg, Vg, LG, Wf, useAsync, out, batch);
}
